// EdgeConv_24756191494465
// MI455X (gfx1250) — compile-verified, matrix operand order checked by reading, not run on hardware
//
#include <hip/hip_runtime.h>
#include <hip/hip_bf16.h>
#include <math.h>

typedef _Float16 h2   __attribute__((ext_vector_type(2)));   // 4 bytes
typedef _Float16 h8   __attribute__((ext_vector_type(8)));   // 16 bytes
typedef _Float16 v16h __attribute__((ext_vector_type(16)));  // 32 bytes (WMMA A/B)
typedef float    v8f  __attribute__((ext_vector_type(8)));   // WMMA C/D

#define NODES 50000
#define KNN   20

// Workspace layout:
//   [0, 6,400,000)           : xT  = f16 transposed x, [NODES][64]
//   [6,400,000, +16,384)     : wsA = f16 W pre-swizzled into WMMA A-operand layout
#define WSA_OFF_BYTES (NODES * 64 * 2)

// ---- Prologue 1: x [64][N] fp32 -> xT [N][64] f16 (contiguous 128B per node column)
__global__ void prep_x_kernel(const float* __restrict__ x, h2* __restrict__ xT) {
  int tid = blockIdx.x * blockDim.x + threadIdx.x;   // one channel-pair per thread
  if (tid >= NODES * 32) return;
  int d = tid & 31, n = tid >> 5;
  h2 v;
  v.x = (_Float16)x[(2 * d) * NODES + n];
  v.y = (_Float16)x[(2 * d + 1) * NODES + n];
  xT[n * 32 + d] = v;                                // one 32-bit store
}

// ---- Prologue 2: W [64][128] fp32 -> f16 in CDNA5 WMMA 16-bit A-matrix layout.
// Tile (kstep,mtile): lane L (L<16 -> M=L, K-half h=0; L>=16 -> M=L-16, h=1),
// VGPR v holds K pair Kl = (v<4 ? 2v : 2v+8) + 8h.
__global__ void prep_w_kernel(const float* __restrict__ W, h2* __restrict__ wsA) {
  int tid = blockIdx.x * blockDim.x + threadIdx.x;
  if (tid >= 4 * 4 * 32) return;
  int lane = tid & 31;
  int tile = tid >> 5;                 // tile = kstep*4 + mtile
  int kstep = tile >> 2, mtile = tile & 3;
  int m = lane & 15, h = lane >> 4;
  int M = mtile * 16 + m;
  #pragma unroll
  for (int v = 0; v < 8; ++v) {
    int Kl = (v < 4 ? 2 * v : 2 * v + 8) + 8 * h;
    int Kg = kstep * 32 + Kl;
    h2 pk;
    pk.x = (_Float16)W[M * 128 + Kg];
    pk.y = (_Float16)W[M * 128 + Kg + 1];
    wsA[(tile * 32 + lane) * 8 + v] = pk;
  }
}

__device__ __forceinline__ v16h mkb(h8 lo, h8 hi) {
  return __builtin_shufflevector(lo, hi, 0,1,2,3,4,5,6,7,8,9,10,11,12,13,14,15);
}

// ---- Main: one wave32 per node. GEMM W[64x128] x Feat[128x32] via WMMA f16.
__global__ __launch_bounds__(256)
void edgeconv_wmma_kernel(const _Float16* __restrict__ xT,
                          const _Float16* __restrict__ wsA,
                          const long long* __restrict__ eidx,
                          const float* __restrict__ dis,
                          const float* __restrict__ bias,
                          const float* __restrict__ att_w,
                          const float* __restrict__ att_b,
                          float* __restrict__ out) {
  const int lane = threadIdx.x & 31;
  const int wave = threadIdx.x >> 5;
  const int node = blockIdx.x * 8 + wave;          // uniform per wave
  if (node >= NODES) return;                        // wave-uniform branch, EXEC stays full
  const int h = lane >> 4, c = lane & 15;

  // lanes 0..19 hold edge k = lane
  int ii = 0, jj = 0;
  float d = 0.f, aw = 0.f;
  if (lane < KNN) {
    jj = (int)eidx[node * KNN + lane];                      // edge_index[0] -> x_j
    ii = (int)eidx[NODES * KNN + node * KNN + lane];        // edge_index[1] -> x_i
    d  = dis[node * KNN + lane];
    aw = att_w[lane];
  }
  // attention scaler: tanh(sum_k att_w[k]*dis[k] + att_b) + 1
  float p = aw * d;
  p += __shfl_xor(p, 16, 32); p += __shfl_xor(p, 8, 32);
  p += __shfl_xor(p, 4, 32);  p += __shfl_xor(p, 2, 32);  p += __shfl_xor(p, 1, 32);
  const float scaler = tanhf(p + att_b[0]) + 1.f;
  // suppression = 2*sigmoid(-d*scaler); 0 for padded columns (k >= 20)
  const float supp = (lane < KNN) ? (2.f / (1.f + __expf(d * scaler))) : 0.f;

  // this lane owns matrix columns c (edge c) and 16+c (edge 16+c)
  const float s0 = __shfl(supp, c, 32);
  const float s1 = __shfl(supp, 16 + c, 32);
  const int i0 = __shfl(ii, c, 32),      j0 = __shfl(jj, c, 32);
  const int i1 = __shfl(ii, 16 + c, 32), j1 = __shfl(jj, 16 + c, 32);   // 0 (clamped) if invalid

  const h8*   X = (const h8*)xT;     // 8 x 16B chunks per node column
  const v16h* A = (const v16h*)wsA;

  // accumulators start at bias; rows m*16+8h..+7 are contiguous -> one v8f load per tile
  v8f acc[4][2];
  #pragma unroll
  for (int m = 0; m < 4; ++m) {
    v8f bv = ((const v8f*)bias)[2 * m + h];
    acc[m][0] = bv;
    acc[m][1] = bv;
  }

  // K-steps: g=0,1 -> rows 0..63 = x_i ; g=2,3 -> rows 64..127 = x_j - x_i
  #pragma unroll
  for (int g = 0; g < 4; ++g) {
    const int cb = 4 * (g & 1) + h;          // chunk base within the 8-chunk column
    v16h B0, B1;
    if (g < 2) {
      B0 = mkb(X[i0 * 8 + cb], X[i0 * 8 + cb + 2]);
      B1 = mkb(X[i1 * 8 + cb], X[i1 * 8 + cb + 2]);
    } else {
      B0 = mkb(X[j0 * 8 + cb] - X[i0 * 8 + cb], X[j0 * 8 + cb + 2] - X[i0 * 8 + cb + 2]);
      B1 = mkb(X[j1 * 8 + cb] - X[i1 * 8 + cb], X[j1 * 8 + cb + 2] - X[i1 * 8 + cb + 2]);
    }
    #pragma unroll
    for (int m = 0; m < 4; ++m) {
      v16h Am = A[(g * 4 + m) * 32 + lane];
      acc[m][0] = __builtin_amdgcn_wmma_f32_16x16x32_f16(false, Am, false, B0,
                                                         (short)0, acc[m][0], false, false);
      acc[m][1] = __builtin_amdgcn_wmma_f32_16x16x32_f16(false, Am, false, B1,
                                                         (short)0, acc[m][1], false, false);
    }
  }

  // Epilogue: relu * suppression, then FOLDING max-reduction over the 16 lanes of
  // each half (columns; padded cols are exactly 0). Each butterfly level keeps the
  // rows this lane will own and exchanges the rest: 8 shuffles per M-tile vs 32.
  const int b3 = (c >> 3) & 1, b2 = (c >> 2) & 1, b1 = (c >> 1) & 1;
  const int rown = b3 * 4 + b2 * 2 + b1;            // row this lane owns at the end
  #pragma unroll
  for (int m = 0; m < 4; ++m) {
    float va[8];
    #pragma unroll
    for (int r = 0; r < 8; ++r) {
      float a0 = fmaxf(acc[m][0][r], 0.f) * s0;
      float a1 = fmaxf(acc[m][1][r], 0.f) * s1;
      va[r] = fmaxf(a0, a1);
    }
    // level 0 (xor 8): keep va[b3*4+t], send va[(1-b3)*4+t]
    float w0 = fmaxf(b3 ? va[4] : va[0], __shfl_xor(b3 ? va[0] : va[4], 8, 32));
    float w1 = fmaxf(b3 ? va[5] : va[1], __shfl_xor(b3 ? va[1] : va[5], 8, 32));
    float w2 = fmaxf(b3 ? va[6] : va[2], __shfl_xor(b3 ? va[2] : va[6], 8, 32));
    float w3 = fmaxf(b3 ? va[7] : va[3], __shfl_xor(b3 ? va[3] : va[7], 8, 32));
    // level 1 (xor 4): keep w[b2*2+t], send w[(1-b2)*2+t]
    float u0 = fmaxf(b2 ? w2 : w0, __shfl_xor(b2 ? w0 : w2, 4, 32));
    float u1 = fmaxf(b2 ? w3 : w1, __shfl_xor(b2 ? w1 : w3, 4, 32));
    // level 2 (xor 2): keep u[b1], send u[1-b1]
    float z = fmaxf(b1 ? u1 : u0, __shfl_xor(b1 ? u0 : u1, 2, 32));
    // level 3 (xor 1): same row in both lanes, plain butterfly
    z = fmaxf(z, __shfl_xor(z, 1, 32));
    if ((c & 1) == 0) out[(m * 16 + 8 * h + rown) * NODES + node] = z;
  }
}

extern "C" void kernel_launch(void* const* d_in, const int* in_sizes, int n_in,
                              void* d_out, int out_size, void* d_ws, size_t ws_size,
                              hipStream_t stream) {
  const float*     x     = (const float*)d_in[0];
  const long long* eidx  = (const long long*)d_in[1];   // int64 edge_index [2,1,N,K]
  const float*     dis   = (const float*)d_in[2];
  const float*     W     = (const float*)d_in[3];
  const float*     b     = (const float*)d_in[4];
  const float*     att_w = (const float*)d_in[5];
  const float*     att_b = (const float*)d_in[6];
  float*           out   = (float*)d_out;

  h2* xT  = (h2*)d_ws;
  h2* wsA = (h2*)((char*)d_ws + WSA_OFF_BYTES);

  prep_x_kernel<<<(NODES * 32 + 255) / 256, 256, 0, stream>>>(x, xT);
  prep_w_kernel<<<2, 256, 0, stream>>>(W, wsA);
  edgeconv_wmma_kernel<<<NODES / 8, 256, 0, stream>>>((const _Float16*)xT, (const _Float16*)wsA,
                                                      eidx, dis, b, att_w, att_b, out);
}
